// BinsEdgeAccuracyLoss_78013785964992
// MI455X (gfx1250) — compile-verified, matrix operand order checked by reading, not run on hardware
//
#include <hip/hip_runtime.h>
#include <hip/hip_bf16.h>

typedef __attribute__((ext_vector_type(4)))  float     v4f;
typedef __attribute__((ext_vector_type(16))) _Float16  v16h;
typedef __attribute__((ext_vector_type(8)))  float     v8f;

#define BLOCK          256
#define V4_PER_THREAD  16       // 16 float4 = 64 elements per thread
#define WS_SLOTS       256

// Zero the workspace accumulator slots each launch (harness does not re-poison).
__global__ void BinsEdge_zero_ws(unsigned* __restrict__ ws) {
    ws[threadIdx.x] = 0u;
}

// Streaming count kernel: bandwidth-bound pass over input/target.
__global__ __launch_bounds__(BLOCK) void BinsEdge_count_kernel(
    const float* __restrict__ inp,
    const float* __restrict__ tgt,
    const float* __restrict__ bins,
    unsigned*    __restrict__ ws,
    long long n)
{
    // Uniform bin grid parameters (bins = linspace(lo, hi, 9)).
    const float lo    = bins[0];
    const float hi    = bins[8];
    const float binw  = (hi - lo) * 0.125f;          // bin width (0.25)
    const float c1    = 1.0f / binw;                 // 4.0
    const float c0    = -lo * c1;                    // 4.0  -> ceil(fma(x,c1,c0)) = bin idx + 1
    const float dstep = (hi - lo) * (1.0f / 7.0f);   // dval spacing 2/7
    const float d1    = 1.0f / dstep;                // 3.5
    const float d0    = -lo * d1 + 1.0f;             // 4.5  -> rint(fma(t,d1,d0)) = target bin + 1

    const v4f* in4 = (const v4f*)inp;
    const v4f* tg4 = (const v4f*)tgt;
    const long long n4   = n >> 2;
    const long long base = (long long)blockIdx.x * (BLOCK * V4_PER_THREAD) + threadIdx.x;

    int cnt = 0;
#pragma unroll
    for (int k = 0; k < V4_PER_THREAD; ++k) {
        long long idx = base + (long long)k * BLOCK;
        if (idx < n4) {
            v4f x = __builtin_nontemporal_load(&in4[idx]);
            v4f t = __builtin_nontemporal_load(&tg4[idx]);
#pragma unroll
            for (int c = 0; c < 4; ++c) {
                float xc = x[c];
                float tc = t[c];
                // bin index of x (right-closed bins); max(.,1) folds the
                // "exactly on lowest edge" case together with xc >= lo.
                float fi = __builtin_fmaxf(
                               __builtin_ceilf(__builtin_fmaf(xc, c1, c0)), 1.0f);
                // bin index implied by target (targets are exact grid values)
                float fj = __builtin_rintf(__builtin_fmaf(tc, d1, d0));
                cnt += (int)((xc >= lo) & (xc <= hi) & (fi == fj));
            }
        }
    }

    // ---- intra-wave (32-lane) reduction via one V_WMMA_F32_16X16X32_F16 ----
    // A[m][0] = cnt(lane m), A[m][8] = cnt(lane m+16), rest 0; B = ones.
    // D[m][n] = cnt(m) + cnt(m+16) for every column n.
    v16h a;
#pragma unroll
    for (int i = 0; i < 16; ++i) a[i] = (_Float16)0.0f;
    a[0] = (_Float16)(float)cnt;            // cnt <= 64, exact in f16

    v16h ones;
#pragma unroll
    for (int i = 0; i < 16; ++i) ones[i] = (_Float16)1.0f;

    v8f acc = {};
    acc = __builtin_amdgcn_wmma_f32_16x16x32_f16(
              /*neg_a=*/false, a, /*neg_b=*/false, ones,
              /*c_mod=*/(short)0, acc, /*reuse_a=*/false, /*reuse_b=*/false);

    // Lane L(<16) holds rows 0..7 of column L; lane L+16 holds rows 8..15.
    float s = acc[0] + acc[1] + acc[2] + acc[3] + acc[4] + acc[5] + acc[6] + acc[7];

    unsigned lane = threadIdx.x & 31u;
    if (lane == 0u || lane == 16u) {
        // lane0 + lane16 partials together = exact wave count (<= 2048, exact f32)
        atomicAdd(&ws[blockIdx.x & (WS_SLOTS - 1)], (unsigned)(s + 0.5f));
    }
}

// Reduce the 256 slots and emit the scalar loss.
__global__ void BinsEdge_finalize(const unsigned* __restrict__ ws,
                                  float* __restrict__ out, float inv_numel) {
    __shared__ unsigned sh[WS_SLOTS];
    sh[threadIdx.x] = ws[threadIdx.x];
    __syncthreads();
    for (int off = WS_SLOTS / 2; off > 0; off >>= 1) {
        if ((int)threadIdx.x < off) sh[threadIdx.x] += sh[threadIdx.x + off];
        __syncthreads();
    }
    if (threadIdx.x == 0) out[0] = 1.0f - (float)sh[0] * inv_numel;
}

extern "C" void kernel_launch(void* const* d_in, const int* in_sizes, int n_in,
                              void* d_out, int out_size, void* d_ws, size_t ws_size,
                              hipStream_t stream) {
    const float* inp  = (const float*)d_in[0];
    const float* tgt  = (const float*)d_in[1];
    const float* bins = (const float*)d_in[2];
    unsigned*    ws   = (unsigned*)d_ws;
    float*       out  = (float*)d_out;

    long long n  = (long long)in_sizes[0];          // 4096*16384 = 67,108,864
    long long n4 = (n + 3) >> 2;
    long long per_block = (long long)BLOCK * V4_PER_THREAD;
    int blocks = (int)((n4 + per_block - 1) / per_block);   // 4096 for reference shape

    BinsEdge_zero_ws<<<1, WS_SLOTS, 0, stream>>>(ws);
    BinsEdge_count_kernel<<<blocks, BLOCK, 0, stream>>>(inp, tgt, bins, ws, n);
    BinsEdge_finalize<<<1, WS_SLOTS, 0, stream>>>(ws, out, 1.0f / (float)n);
}
